// Model_29059748725031
// MI455X (gfx1250) — matrix operand order repaired and checked by reading, not compiled, not run on hardware
//
#include <hip/hip_runtime.h>
#include <hip/hip_bf16.h>

typedef __attribute__((ext_vector_type(16))) _Float16 v16h;
typedef __attribute__((ext_vector_type(8)))  _Float16 v8h;
typedef __attribute__((ext_vector_type(8)))  float    v8f;
typedef __attribute__((ext_vector_type(4)))  int      v4i;

#define EPSV 1e-6f
#define SCALV 5.0f

#if defined(__has_builtin)
#if __has_builtin(__builtin_amdgcn_global_load_async_to_lds_b128)
#define HAVE_ASYNC_LDS 1
#endif
#endif

#ifdef HAVE_ASYNC_LDS
// src must point into global memory; dst must be an LDS (shared) location.
__device__ __forceinline__ void async_copy16(const _Float16* src, _Float16* dst_lds) {
  __builtin_amdgcn_global_load_async_to_lds_b128(
      (__attribute__((address_space(1))) v4i*)(unsigned long long)(uintptr_t)src,
      (__attribute__((address_space(3))) v4i*)dst_lds,
      0, 0);
}
#endif

// ---------------- workspace layout (bytes) ----------------
// All intermediate activations: f16 NHWC (channel stride = padded channels).
static constexpr size_t O_BUF1  = 0;                 // h1 / t1 / enc    (16.8 MB f16, ch-stride 64)
static constexpr size_t O_BUF2  = 16777216;          // h2 / fxs / dh1   (16.8 MB f16)
static constexpr size_t O_PA    = 33554432;          // shape parts f32 NCHW (8.4 MB)
static constexpr size_t O_PB    = 41943040;          // app parts   f32 NCHW (8.4 MB)
static constexpr size_t O_DH2   = 50331648;          // decoder h2 f16 NHWC 128x128x32 (33.6 MB)
static constexpr size_t O_ALPHA = 83886080;          // 32*16*32 f32
static constexpr size_t O_MU    = 83951616;
static constexpr size_t O_LINV  = 83955712;
static constexpr size_t O_MUA   = 83963904;
static constexpr size_t O_LINVA = 83968000;
static constexpr size_t O_HSUM  = 83976192;
static constexpr size_t O_ACC   = 83978240;
static constexpr size_t O_WPACK = 83978304;          // f16 packed weights (16B aligned)
// packed-weight sub-offsets (halves): layout [tap][co][ciPad64]
static constexpr size_t WP_ES2 = 0;       // 9*64*64
static constexpr size_t WP_ESP = 36864;   // 9*16*64
static constexpr size_t WP_EA1 = 46080;   // 9*64*64
static constexpr size_t WP_EA2 = 82944;   // 9*32*64
static constexpr size_t WP_D1  = 101376;  // 9*64*64
static constexpr size_t WP_D2  = 138240;  // 9*32*64

// ---------------- wave32 reductions ----------------
__device__ __forceinline__ float wsum(float v) {
  #pragma unroll
  for (int o = 16; o; o >>= 1) v += __shfl_down(v, o);
  return v;
}
__device__ __forceinline__ float wmaxf(float v) {
  #pragma unroll
  for (int o = 16; o; o >>= 1) v = fmaxf(v, __shfl_down(v, o));
  return v;
}
__device__ __forceinline__ float blockSum(float v, float* sm, float* bc) {
  int lane = threadIdx.x & 31, w = threadIdx.x >> 5, nw = blockDim.x >> 5;
  v = wsum(v);
  if (lane == 0) sm[w] = v;
  __syncthreads();
  if (threadIdx.x == 0) { float s = 0.f; for (int i = 0; i < nw; ++i) s += sm[i]; *bc = s; }
  __syncthreads();
  return *bc;
}
__device__ __forceinline__ float blockMax(float v, float* sm, float* bc) {
  int lane = threadIdx.x & 31, w = threadIdx.x >> 5, nw = blockDim.x >> 5;
  v = wmaxf(v);
  if (lane == 0) sm[w] = v;
  __syncthreads();
  if (threadIdx.x == 0) { float s = -1e30f; for (int i = 0; i < nw; ++i) s = fmaxf(s, sm[i]); *bc = s; }
  __syncthreads();
  return *bc;
}

// ---------------- weight pack: OIHW f32 -> [tap][co][ciPad] f16 ----------------
__global__ void pack_weights(const float* __restrict__ w, _Float16* __restrict__ wp,
                             int CIN, int CINP, int COUT) {
  int i = blockIdx.x * blockDim.x + threadIdx.x;
  int total = 9 * COUT * CINP;
  if (i >= total) return;
  int ci = i % CINP;
  int co = (i / CINP) % COUT;
  int tap = i / (CINP * COUT);
  float v = (ci < CIN) ? w[(co * CIN + ci) * 9 + tap] : 0.0f;
  wp[i] = (_Float16)v;
}

// ---------------- stride-2 3x3 conv (NCHW f32 image -> NHWC f16) + relu ----------------
__global__ void conv_s2_relu(const float* __restrict__ in, const float* __restrict__ w,
                             const float* __restrict__ bias, _Float16* __restrict__ out) {
  int idx = blockIdx.x * blockDim.x + threadIdx.x;   // ((b*4096 + y*64+x)*64 + co)
  int co = idx & 63, x = (idx >> 6) & 63, y = (idx >> 12) & 63, b = idx >> 18;
  float acc = bias[co];
  for (int ci = 0; ci < 3; ++ci)
    for (int r = 0; r < 3; ++r) {
      int iy = 2 * y + r; if (iy >= 128) continue;
      for (int s = 0; s < 3; ++s) {
        int ix = 2 * x + s; if (ix >= 128) continue;
        acc += w[((co * 3 + ci) * 3 + r) * 3 + s] * in[((size_t)(b * 3 + ci) * 128 + iy) * 128 + ix];
      }
    }
  out[idx] = (_Float16)fmaxf(acc, 0.0f);
}

// ---------------- implicit-GEMM 3x3 conv, WMMA f16, NHWC f16 in ----------------
// One wave = 16 Cout x 64 pixels. LDS slab staged with async-to-LDS b128 when available.
// ACT: 0=none, 1=relu. UP: nearest x2 upsample folded into addressing.
// OUTF32: 1 -> write f32 NCHW (parts path); 0 -> f16 NHWC (stride COUT) via b128 stores.
template <int CINP, int COUT, int ACT, int UP, int OUTF32>
__global__ __launch_bounds__(32 * (COUT / 16))
void conv3x3_wmma(const _Float16* __restrict__ in, const _Float16* __restrict__ wp,
                  const float* __restrict__ bias, void* __restrict__ outv, int H, int W) {
  constexpr int NCOT = COUT / 16;
  constexpr int CH8  = CINP / 8;                 // b128 chunks per pixel
  __shared__ __align__(16) _Float16 slab[3][66][CINP];
  int strips = W >> 6;
  int y  = blockIdx.x / strips;
  int x0 = (blockIdx.x % strips) << 6;
  int b  = blockIdx.y;
  int HI = UP ? (H >> 1) : H;
  int WI = UP ? (W >> 1) : W;
  int tid = threadIdx.x;

  __builtin_prefetch(wp + (size_t)(tid & 31) * 64, 0, 1);  // warm packed weights

  // stage slab: rows y-1..y+1, cols x0-1..x0+64, CINP channels, in 16-byte chunks
  const int chunksPerRow = 66 * CH8;
  const int totalChunks  = 3 * chunksPerRow;
  for (int c = tid; c < totalChunks; c += 32 * NCOT) {
    int r   = c / chunksPerRow;
    int rem = c - r * chunksPerRow;
    int col = rem / CH8;
    int k8  = rem - col * CH8;
    int yy = y - 1 + r;
    int xx = x0 - 1 + col;
    _Float16* dst = ((_Float16*)slab) + (size_t)c * 8;
    bool valid = (yy >= 0 && yy < H && xx >= 0 && xx < W);
    int sy = UP ? (yy >> 1) : yy;
    int sx = UP ? (xx >> 1) : xx;
    const _Float16* src = in + (((size_t)b * HI + sy) * WI + sx) * CINP + k8 * 8;
#ifdef HAVE_ASYNC_LDS
    if (valid) {
      async_copy16(src, dst);
    } else {
      v8h z;
      #pragma unroll
      for (int e = 0; e < 8; ++e) z[e] = (_Float16)0.0f;
      *(v8h*)dst = z;
    }
#else
    v8h v;
    if (valid) {
      v = *(const v8h*)src;
    } else {
      #pragma unroll
      for (int e = 0; e < 8; ++e) v[e] = (_Float16)0.0f;
    }
    *(v8h*)dst = v;
#endif
  }
#ifdef HAVE_ASYNC_LDS
#if __has_builtin(__builtin_amdgcn_s_wait_asynccnt)
  __builtin_amdgcn_s_wait_asynccnt(0);
#else
  asm volatile("s_wait_asynccnt 0x0" ::: "memory");
#endif
#endif
  __syncthreads();

  int lane = tid & 31;
  int wv   = tid >> 5;        // cout tile
  int m    = lane & 15;
  int hi   = lane >> 4;

  v8f acc[4];
  #pragma unroll
  for (int nt = 0; nt < 4; ++nt)
    #pragma unroll
    for (int r = 0; r < 8; ++r) acc[nt][r] = 0.0f;

  for (int tap = 0; tap < 9; ++tap) {
    int dy  = tap / 3;
    int dxo = tap % 3;
    #pragma unroll
    for (int kc = 0; kc < CINP / 32; ++kc) {
      // A fragment: lane<16 holds K {0..7,16..23}; lane>=16 holds K {8..15,24..31}
      const _Float16* wpb = wp + ((size_t)(tap * COUT + wv * 16 + m)) * CINP + kc * 32 + hi * 8;
      v8h a0 = *(const v8h*)wpb;
      v8h a1 = *(const v8h*)(wpb + 16);
      v16h A;
      #pragma unroll
      for (int e = 0; e < 8; ++e) { A[e] = a0[e]; A[e + 8] = a1[e]; }
      #pragma unroll
      for (int nt = 0; nt < 4; ++nt) {
        // B fragment: lane<16 holds K {0..7,16..23}; lane>=16 holds K {8..15,24..31}
        int col = nt * 16 + m + dxo;
        const _Float16* sb = &slab[dy][col][kc * 32 + hi * 8];
        v8h b0 = *(const v8h*)sb;
        v8h b1 = *(const v8h*)(sb + 16);
        v16h Bv;
        #pragma unroll
        for (int e = 0; e < 8; ++e) { Bv[e] = b0[e]; Bv[e + 8] = b1[e]; }
        acc[nt] = __builtin_amdgcn_wmma_f32_16x16x32_f16(
            false, A, false, Bv, (short)0, acc[nt], false, false);
      }
    }
  }

  // epilogue: C layout -> lane holds 8 consecutive Cout (wv*16 + hi*8 + r), pixel n = lane&15
  if constexpr (OUTF32) {
    float* out = (float*)outv;
    #pragma unroll
    for (int nt = 0; nt < 4; ++nt)
      #pragma unroll
      for (int r = 0; r < 8; ++r) {
        int mm = hi * 8 + r;
        float v = acc[nt][r] + bias[wv * 16 + mm];
        if (ACT == 1) v = fmaxf(v, 0.0f);
        out[((size_t)(b * COUT + wv * 16 + mm) * H + y) * W + x0 + nt * 16 + m] = v;
      }
  } else {
    _Float16* out = (_Float16*)outv;
    #pragma unroll
    for (int nt = 0; nt < 4; ++nt) {
      v8h o;
      #pragma unroll
      for (int r = 0; r < 8; ++r) {
        float v = acc[nt][r] + bias[wv * 16 + hi * 8 + r];
        if (ACT == 1) v = fmaxf(v, 0.0f);
        o[r] = (_Float16)v;
      }
      *(v8h*)(out + (((size_t)b * H + y) * W + x0 + nt * 16 + m) * COUT + wv * 16 + hi * 8) = o;
    }
  }
}

// ---------------- spatial softmax over 4096, in place (f32 NCHW parts) ----------------
__global__ void softmax4096(float* __restrict__ p) {
  __shared__ float sm[8];
  __shared__ float bc;
  float* base = p + (size_t)blockIdx.x * 4096;
  int t = threadIdx.x;
  float v[16];
  float mx = -1e30f;
  #pragma unroll
  for (int j = 0; j < 16; ++j) { v[j] = base[j * 256 + t]; mx = fmaxf(mx, v[j]); }
  mx = blockMax(mx, sm, &bc);
  float s = 0.f;
  #pragma unroll
  for (int j = 0; j < 16; ++j) { v[j] = __expf(v[j] - mx); s += v[j]; }
  s = blockSum(s, sm, &bc);
  float inv = 1.0f / s;
  #pragma unroll
  for (int j = 0; j < 16; ++j) base[j * 256 + t] = v[j] * inv;
}

// ---------------- soft-argmax mu + 2x2 cov -> L_inv ----------------
__global__ void mu_cov(const float* __restrict__ parts, float* __restrict__ mu,
                       float* __restrict__ linv) {
  __shared__ float sm[8];
  __shared__ float bc;
  int bk = blockIdx.x;
  const float* p = parts + (size_t)bk * 4096;
  int t = threadIdx.x;
  float Sy = 0, Sx = 0, Syy = 0, Syx = 0, Sxx = 0;
  #pragma unroll
  for (int j = 0; j < 16; ++j) {
    int hw = j * 256 + t;
    float pv = p[hw];
    float gy = -1.0f + (float)(hw >> 6) * (2.0f / 63.0f);
    float gx = -1.0f + (float)(hw & 63) * (2.0f / 63.0f);
    Sy += pv * gy; Sx += pv * gx;
    Syy += pv * gy * gy; Syx += pv * gy * gx; Sxx += pv * gx * gx;
  }
  Sy = blockSum(Sy, sm, &bc);
  Sx = blockSum(Sx, sm, &bc);
  Syy = blockSum(Syy, sm, &bc);
  Syx = blockSum(Syx, sm, &bc);
  Sxx = blockSum(Sxx, sm, &bc);
  if (t == 0) {
    float c00 = Syy - Sy * Sy, c01 = Syx - Sy * Sx, c11 = Sxx - Sx * Sx;
    float a  = sqrtf(c00 + EPSV);
    float bb = c01 / (a + EPSV);
    float c  = sqrtf(fmaxf(c11 - bb * bb, 0.0f) + EPSV);
    float s  = SCALV / (a * c + EPSV);
    mu[bk * 2 + 0] = Sy;
    mu[bk * 2 + 1] = Sx;
    linv[bk * 4 + 0] = s * c;  linv[bk * 4 + 1] = 0.0f;
    linv[bk * 4 + 2] = -s * bb; linv[bk * 4 + 3] = s * a;
  }
}

// ---- alpha[b,k,f] = sum_hw fxs[b,hw,f] * parts[b,k,hw]  (fxs f16 NHWC, stride 32) ----
__global__ void alpha_kernel(const float* __restrict__ parts, const _Float16* __restrict__ fxs,
                             float* __restrict__ alpha) {
  __shared__ float accf[32];
  int bk = blockIdx.x, b = bk >> 4, t = threadIdx.x;
  if (t < 32) accf[t] = 0.0f;
  __syncthreads();
  float part[32];
  #pragma unroll
  for (int f = 0; f < 32; ++f) part[f] = 0.0f;
  const float* pp = parts + (size_t)bk * 4096;
  const _Float16* fb = fxs + (size_t)b * 4096 * 32;
  for (int hw = t; hw < 4096; hw += 256) {
    float p = pp[hw];
    const v8h* f8 = (const v8h*)(fb + (size_t)hw * 32);
    #pragma unroll
    for (int c = 0; c < 4; ++c) {
      v8h v = f8[c];
      #pragma unroll
      for (int e = 0; e < 8; ++e) part[c * 8 + e] += p * (float)v[e];
    }
  }
  #pragma unroll
  for (int f = 0; f < 32; ++f) atomicAdd(&accf[f], part[f]);
  __syncthreads();
  if (t < 32) alpha[bk * 32 + t] = accf[t];
}

// ---- Cauchy heatmap -> enc channel k (f16 NHWC, stride 64) + per-part sums ----
__global__ void render_heat(const float* __restrict__ mu, const float* __restrict__ linv,
                            _Float16* __restrict__ enc, float* __restrict__ hsum) {
  __shared__ float sm[8];
  __shared__ float bc;
  int bk = blockIdx.x, b = bk >> 4, k = bk & 15;
  float m0 = mu[bk * 2], m1 = mu[bk * 2 + 1];
  float L00 = linv[bk * 4], L01 = linv[bk * 4 + 1], L10 = linv[bk * 4 + 2], L11 = linv[bk * 4 + 3];
  _Float16* dst = enc + (size_t)b * 4096 * 64 + k;
  int t = threadIdx.x;
  float s = 0.f;
  #pragma unroll
  for (int j = 0; j < 16; ++j) {
    int hw = j * 256 + t;
    float gy = -1.0f + (float)(hw >> 6) * (2.0f / 63.0f);
    float gx = -1.0f + (float)(hw & 63) * (2.0f / 63.0f);
    float d0 = gy - m0, d1 = gx - m1;
    float p0 = d0 * L00 + d1 * L10;
    float p1 = d0 * L01 + d1 * L11;
    float h = 1.0f / (1.0f + p0 * p0 + p1 * p1);
    dst[(size_t)hw * 64] = (_Float16)h;
    s += h;
  }
  s = blockSum(s, sm, &bc);
  if (t == 0) hsum[bk] = s;
}

// ---- fmap: enc ch 16+f = sum_k alpha[b,k,f]*heat_norm; zero pad ch 48..63 ----
__global__ void fmap_kernel(const float* __restrict__ alpha, const float* __restrict__ hsum,
                            _Float16* __restrict__ enc) {
  __shared__ float al[512];
  __shared__ float hinv[16];
  int b = blockIdx.y, t = threadIdx.x;
  int hw = blockIdx.x * 256 + t;
  if (t < 16) hinv[t] = 1.0f / (hsum[b * 16 + t] + EPSV);
  for (int i = t; i < 512; i += 256) al[i] = alpha[b * 512 + i];
  __syncthreads();
  _Float16* base = enc + ((size_t)b * 4096 + hw) * 64;
  float h[16];
  #pragma unroll
  for (int k = 0; k < 16; ++k) h[k] = (float)base[k] * hinv[k];
  #pragma unroll
  for (int f = 0; f < 32; ++f) {
    float a = 0.f;
    #pragma unroll
    for (int k = 0; k < 16; ++k) a += al[k * 32 + f] * h[k];
    base[16 + f] = (_Float16)a;
  }
  #pragma unroll
  for (int z = 48; z < 64; ++z) base[z] = (_Float16)0.0f;
}

// ---- final 3x3 conv 32->3 + sigmoid @128x128, NHWC f16 in, NCHW f32 out ----
__global__ void conv_final_sigmoid(const _Float16* __restrict__ in, const float* __restrict__ w,
                                   const float* __restrict__ bias, float* __restrict__ out) {
  __shared__ _Float16 tile[18][18][32];
  __shared__ float wl[864];
  int b = blockIdx.y;
  int ty = (blockIdx.x >> 3) * 16, tx = (blockIdx.x & 7) * 16;
  int t = threadIdx.x;
  for (int i = t; i < 864; i += 256) wl[i] = w[i];
  for (int c = t; c < 18 * 18 * 4; c += 256) {      // 16-byte chunks: (pix, k8)
    int k8 = c & 3, pix = c >> 2;
    int yy = pix / 18, xx = pix % 18;
    int gy = ty - 1 + yy, gx = tx - 1 + xx;
    v8h v;
    if (gy >= 0 && gy < 128 && gx >= 0 && gx < 128) {
      v = *(const v8h*)(in + (((size_t)b * 128 + gy) * 128 + gx) * 32 + k8 * 8);
    } else {
      #pragma unroll
      for (int e = 0; e < 8; ++e) v[e] = (_Float16)0.0f;
    }
    *(v8h*)(&tile[yy][xx][k8 * 8]) = v;
  }
  __syncthreads();
  int ly = t >> 4, lx = t & 15;
  float a0 = bias[0], a1 = bias[1], a2 = bias[2];
  for (int ci = 0; ci < 32; ++ci)
    #pragma unroll
    for (int r = 0; r < 3; ++r)
      #pragma unroll
      for (int s = 0; s < 3; ++s) {
        float v = (float)tile[ly + r][lx + s][ci];
        int wi = (ci * 3 + r) * 3 + s;
        a0 += wl[wi] * v;
        a1 += wl[288 + wi] * v;
        a2 += wl[576 + wi] * v;
      }
  int y = ty + ly, x = tx + lx;
  out[((size_t)(b * 3 + 0) * 128 + y) * 128 + x] = 1.0f / (1.0f + __expf(-a0));
  out[((size_t)(b * 3 + 1) * 128 + y) * 128 + x] = 1.0f / (1.0f + __expf(-a1));
  out[((size_t)(b * 3 + 2) * 128 + y) * 128 + x] = 1.0f / (1.0f + __expf(-a2));
}

// ---------------- losses ----------------
__global__ void zero8(float* a) { if (threadIdx.x < 8) a[threadIdx.x] = 0.0f; }

__global__ void rec_loss_kernel(const float* __restrict__ x, const float* __restrict__ rec,
                                float* __restrict__ acc) {
  __shared__ float sm[8];
  __shared__ float bc;
  int i = blockIdx.x * 1024 + threadIdx.x;
  float s = 0.f;
  #pragma unroll
  for (int j = 0; j < 4; ++j) {
    int idx = i + j * 256;
    float d = x[idx] - rec[idx];
    s += d * d;
  }
  s = blockSum(s, sm, &bc);
  if (threadIdx.x == 0) atomicAdd(acc, s);
}

__global__ void final_loss(const float* __restrict__ mu, const float* __restrict__ linv,
                           const float* __restrict__ mua, const float* __restrict__ linva,
                           const float* __restrict__ coord, const float* __restrict__ vec,
                           const float* __restrict__ acc, float* __restrict__ out_loss) {
  __shared__ float sm[8];
  __shared__ float bc;
  int t = threadIdx.x;
  float s1 = 0, s2 = 0, s3 = 0;
  for (int i = t; i < 1024; i += 256) {
    float d = mu[i] - mua[i];
    s1 += d * d;
    float dd = mua[i] - (coord[i] + vec[i]);
    s3 += dd * dd;
  }
  for (int i = t; i < 2048; i += 256) {
    float d = linv[i] - linva[i];
    s2 += d * d;
  }
  s1 = blockSum(s1, sm, &bc);
  s2 = blockSum(s2, sm, &bc);
  s3 = blockSum(s3, sm, &bc);
  if (t == 0)
    out_loss[0] = acc[0] / 1572864.0f + s1 / 1024.0f + 0.01f * s2 / 2048.0f + s3 / 1024.0f;
}

// ---------------- orchestration ----------------
extern "C" void kernel_launch(void* const* d_in, const int* in_sizes, int n_in,
                              void* d_out, int out_size, void* d_ws, size_t ws_size,
                              hipStream_t stream) {
  (void)in_sizes; (void)n_in; (void)out_size; (void)ws_size;
  const float* x     = (const float*)d_in[0];
  const float* xs    = (const float*)d_in[1];   // x_spatial_transform
  const float* xa    = (const float*)d_in[2];   // x_appearance_transform
  const float* coord = (const float*)d_in[3];
  const float* vec   = (const float*)d_in[4];
  const float* es_w1 = (const float*)d_in[5];  const float* es_b1 = (const float*)d_in[6];
  const float* es_w2 = (const float*)d_in[7];  const float* es_b2 = (const float*)d_in[8];
  const float* es_wp = (const float*)d_in[9];  const float* es_bp = (const float*)d_in[10];
  const float* ea_w1 = (const float*)d_in[11]; const float* ea_b1 = (const float*)d_in[12];
  const float* ea_w2 = (const float*)d_in[13]; const float* ea_b2 = (const float*)d_in[14];
  const float* d_w1  = (const float*)d_in[15]; const float* d_b1  = (const float*)d_in[16];
  const float* d_w2  = (const float*)d_in[17]; const float* d_b2  = (const float*)d_in[18];
  const float* d_w3  = (const float*)d_in[19]; const float* d_b3  = (const float*)d_in[20];
  float* out = (float*)d_out;
  char*  wsb = (char*)d_ws;

  _Float16* buf1 = (_Float16*)(wsb + O_BUF1);
  _Float16* buf2 = (_Float16*)(wsb + O_BUF2);
  float* pA    = (float*)(wsb + O_PA);
  float* pB    = (float*)(wsb + O_PB);
  _Float16* dh2 = (_Float16*)(wsb + O_DH2);
  float* alpB  = (float*)(wsb + O_ALPHA);
  float* muS   = (float*)(wsb + O_MU);
  float* linvS = (float*)(wsb + O_LINV);
  float* muA   = (float*)(wsb + O_MUA);
  float* linvA = (float*)(wsb + O_LINVA);
  float* hsum  = (float*)(wsb + O_HSUM);
  float* acc   = (float*)(wsb + O_ACC);
  _Float16* wpk = (_Float16*)(wsb + O_WPACK);

  // pack conv weights to f16 [tap][co][ciPad64]
  pack_weights<<<(9 * 64 * 64 + 255) / 256, 256, 0, stream>>>(es_w2, wpk + WP_ES2, 64, 64, 64);
  pack_weights<<<(9 * 16 * 64 + 255) / 256, 256, 0, stream>>>(es_wp, wpk + WP_ESP, 64, 64, 16);
  pack_weights<<<(9 * 64 * 64 + 255) / 256, 256, 0, stream>>>(ea_w1, wpk + WP_EA1, 64, 64, 64);
  pack_weights<<<(9 * 32 * 64 + 255) / 256, 256, 0, stream>>>(ea_w2, wpk + WP_EA2, 64, 64, 32);
  pack_weights<<<(9 * 64 * 64 + 255) / 256, 256, 0, stream>>>(d_w1,  wpk + WP_D1,  48, 64, 64);
  pack_weights<<<(9 * 32 * 64 + 255) / 256, 256, 0, stream>>>(d_w2,  wpk + WP_D2,  64, 64, 32);

  dim3 g64(64, 32);    // 64x64 outputs: 1 strip x 64 rows, batch 32
  dim3 g128(256, 32);  // 128x128 outputs: 2 strips x 128 rows

  // ----- shape stream (appearance-transformed image -> geometry) -----
  conv_s2_relu<<<32768, 256, 0, stream>>>(xa, es_w1, es_b1, buf1);
  conv3x3_wmma<64, 64, 1, 0, 0><<<g64, 128, 0, stream>>>(buf1, wpk + WP_ES2, es_b2, buf2, 64, 64);
  conv3x3_wmma<64, 16, 0, 0, 1><<<g64, 32, 0, stream>>>(buf2, wpk + WP_ESP, es_bp, pA, 64, 64);
  softmax4096<<<512, 256, 0, stream>>>(pA);
  mu_cov<<<512, 256, 0, stream>>>(pA, muS, linvS);

  // ----- appearance stream (spatially-transformed image) -----
  conv_s2_relu<<<32768, 256, 0, stream>>>(xs, es_w1, es_b1, buf1);
  conv3x3_wmma<64, 64, 1, 0, 0><<<g64, 128, 0, stream>>>(buf1, wpk + WP_ES2, es_b2, buf2, 64, 64);
  conv3x3_wmma<64, 16, 0, 0, 1><<<g64, 32, 0, stream>>>(buf2, wpk + WP_ESP, es_bp, pB, 64, 64);
  softmax4096<<<512, 256, 0, stream>>>(pB);
  mu_cov<<<512, 256, 0, stream>>>(pB, muA, linvA);

  // e_alpha: buf2 (app_sum) -> buf1 (relu, 64ch) -> buf2 (fxs, 32ch NHWC)
  conv3x3_wmma<64, 64, 1, 0, 0><<<g64, 128, 0, stream>>>(buf2, wpk + WP_EA1, ea_b1, buf1, 64, 64);
  conv3x3_wmma<64, 32, 0, 0, 0><<<g64, 64, 0, stream>>>(buf1, wpk + WP_EA2, ea_b2, buf2, 64, 64);

  // gather part appearances
  alpha_kernel<<<512, 256, 0, stream>>>(pB, buf2, alpB);

  // encoding into buf1 (NHWC, 64ch: 16 heat + 32 fmap + 16 zero pad)
  render_heat<<<512, 256, 0, stream>>>(muS, linvS, buf1, hsum);
  fmap_kernel<<<dim3(16, 32), 256, 0, stream>>>(alpB, hsum, buf1);

  // decoder
  conv3x3_wmma<64, 64, 1, 0, 0><<<g64, 128, 0, stream>>>(buf1, wpk + WP_D1, d_b1, buf2, 64, 64);
  conv3x3_wmma<64, 32, 1, 1, 0><<<g128, 64, 0, stream>>>(buf2, wpk + WP_D2, d_b2, dh2, 128, 128);
  conv_final_sigmoid<<<dim3(64, 32), 256, 0, stream>>>(dh2, d_w3, d_b3, out);

  // losses -> out[1572864]
  zero8<<<1, 32, 0, stream>>>(acc);
  rec_loss_kernel<<<1536, 256, 0, stream>>>(x, out, acc);
  final_loss<<<1, 256, 0, stream>>>(muS, linvS, muA, linvA, coord, vec, acc, out + 1572864);
}
